// DLI_loss_full_6614249636354
// MI455X (gfx1250) — matrix operand order repaired and checked by reading, not compiled, not run on hardware
//
#include <hip/hip_runtime.h>
#include <hip/hip_bf16.h>

typedef __attribute__((ext_vector_type(16))) __bf16 v16bf;
typedef __attribute__((ext_vector_type(8)))  __bf16 v8bf;
typedef __attribute__((ext_vector_type(4)))  __bf16 v4bf;
typedef __attribute__((ext_vector_type(8)))  float  v8f;
typedef __attribute__((ext_vector_type(2)))  float  v2f;

#define BB 32
#define SS 2048
#define DD 1024
#define HH 512
#define TT 64
#define G4 2048   // 4*HH

// ---------------------------------------------------------------------------
// Kernel 1: segment means -> states (B,T,D), plus bp[b,t] = states . W_b
// One block per (b,t). 256 threads x float4 = 1024 columns. Pure streaming,
// this kernel carries the dominant 256MB of HBM traffic.
// ---------------------------------------------------------------------------
__global__ void k_states_bp(const float* __restrict__ enc,
                            const int*  __restrict__ ends,
                            const int*  __restrict__ tlen,
                            const float* __restrict__ Wfc,
                            float* __restrict__ states,
                            float* __restrict__ bp)
{
    const int blk = blockIdx.x;          // b*T + t
    const int b = blk >> 6;
    const int t = blk & 63;
    const int tid = threadIdx.x;

    const int end   = ends[b * TT + t];
    const int start = (t == 0) ? 0 : (ends[b * TT + t - 1] + 1);
    const float inv = 1.0f / (float)(end - start + 1);
    const bool  live = (t < tlen[b]);

    const int col = tid * 4;
    const float* base = enc + ((size_t)b * SS + start) * DD + col;
    float4 acc = make_float4(0.f, 0.f, 0.f, 0.f);
    for (int r = start; r <= end; ++r) {
        __builtin_prefetch(base + 8 * DD, 0, 1);   // global_prefetch_b8
        float4 v = *(const float4*)base;
        acc.x += v.x; acc.y += v.y; acc.z += v.z; acc.w += v.w;
        base += DD;
    }
    float4 st;
    st.x = live ? acc.x * inv : 0.f;
    st.y = live ? acc.y * inv : 0.f;
    st.z = live ? acc.z * inv : 0.f;
    st.w = live ? acc.w * inv : 0.f;
    *(float4*)(states + (size_t)blk * DD + col) = st;

    // bp = states . W_b  (W_b = Wfc[H .. H+D))
    const float* Wb = Wfc + HH;
    float d = st.x * Wb[col] + st.y * Wb[col + 1] + st.z * Wb[col + 2] + st.w * Wb[col + 3];
    __shared__ float red[256];
    red[tid] = d;
    __syncthreads();
    for (int s = 128; s > 0; s >>= 1) {
        if (tid < s) red[tid] += red[tid + s];
        __syncthreads();
    }
    if (tid == 0) bp[blk] = red[0];
}

// ---------------------------------------------------------------------------
// Kernel 2: xW = states @ W_ih.T + b_ih + b_hh   (M=2048, N=2048, K=1024)
// bf16 WMMA, f32 accumulate. Block = 1 mtile x 4 ntiles (4 waves).
// Coalesced b128 global loads stage f32->bf16 tiles in LDS; fragments are
// fetched with ds_load_b128 (A frag = 2x b128, B frag = 2x b128 per the
// CDNA5 16-bit WMMA VGPR layout).
// ---------------------------------------------------------------------------
__global__ void __launch_bounds__(128)
k_xw_gemm(const float* __restrict__ states,
          const float* __restrict__ Wih,
          const float* __restrict__ bih,
          const float* __restrict__ bhh,
          float* __restrict__ xW)
{
    const int mtile = blockIdx.x >> 5;         // 0..127
    const int ngrp  = blockIdx.x & 31;         // 0..31, 64 cols each
    const int wave  = threadIdx.x >> 5;
    const int lane  = threadIdx.x & 31;
    const int half  = lane >> 4;
    const int l15   = lane & 15;
    const int n0    = ngrp * 64 + wave * 16;   // this wave's 16 output cols

    __shared__ __attribute__((aligned(16))) __bf16 Al[16][32];
    __shared__ __attribute__((aligned(16))) __bf16 Bl[4][16][32];

    union { v16bf v; v8bf h[2]; } af, bfr;
    v8f acc = {};
    for (int k0 = 0; k0 < DD; k0 += 32) {
        // ---- stage A tile (16x32 f32 -> bf16), 128 threads x one float4 ----
        {
            const int e = threadIdx.x * 4;     // 0..508
            const int r = e >> 5, c = e & 31;
            float4 v = *(const float4*)&states[(size_t)(mtile * 16 + r) * DD + k0 + c];
            v4bf p; p[0] = (__bf16)v.x; p[1] = (__bf16)v.y; p[2] = (__bf16)v.z; p[3] = (__bf16)v.w;
            *(v4bf*)&Al[r][c] = p;             // ds_store_b64
        }
        // ---- stage B tile per wave (16x32 f32 -> bf16), 32 lanes x 4 float4 ----
#pragma unroll
        for (int q = 0; q < 4; ++q) {
            const int e = lane * 16 + q * 4;   // 0..508
            const int r = e >> 5, c = e & 31;
            float4 v = *(const float4*)&Wih[(size_t)(n0 + r) * DD + k0 + c];
            v4bf p; p[0] = (__bf16)v.x; p[1] = (__bf16)v.y; p[2] = (__bf16)v.z; p[3] = (__bf16)v.w;
            *(v4bf*)&Bl[wave][r][c] = p;
        }
        __syncthreads();
        // ---- fragment fetch: contiguous per CDNA5 layout -> ds_load_b128 ----
        af.h[0]  = *(const v8bf*)&Al[l15][half * 8];            // K = half*8 .. +7
        af.h[1]  = *(const v8bf*)&Al[l15][16 + half * 8];       // K = 16+half*8 .. +7
        bfr.h[0] = *(const v8bf*)&Bl[wave][l15][half * 8];      // K = half*8 .. +7
        bfr.h[1] = *(const v8bf*)&Bl[wave][l15][16 + half * 8]; // K = 16+half*8 .. +7
        acc = __builtin_amdgcn_wmma_f32_16x16x32_bf16(
                  false, af.v, false, bfr.v, (short)0, acc, false, false);
        __syncthreads();
    }
#pragma unroll
    for (int r = 0; r < 8; ++r) {
        const int m = mtile * 16 + (half ? (8 + r) : r);
        const int n = n0 + l15;
        xW[(size_t)m * G4 + n] = acc[r] + bih[n] + bhh[n];
    }
}

// ---------------------------------------------------------------------------
// Kernel 3: one LSTM step. gates = xW[:,t,:] + h_in @ W_hh.T  (f32 WMMA,
// full precision in the recurrence). h tile staged once in LDS (32KB),
// A frag = one ds_load_b64, B frag = contiguous global_load_b64 from
// L2-resident W_hh. grid = 2 mtiles x 32 jtiles; wave g computes gate g.
// ---------------------------------------------------------------------------
__global__ void __launch_bounds__(128)
k_lstm_step(const float* __restrict__ xW,
            const float* __restrict__ Whh,
            const float* __restrict__ Wfc,
            const float* __restrict__ hin,
            float* __restrict__ hout,
            float* __restrict__ cst,
            float* __restrict__ aacc,
            int t)
{
    const int mtile = blockIdx.x >> 5;
    const int jtile = blockIdx.x & 31;
    const int gate  = threadIdx.x >> 5;
    const int lane  = threadIdx.x & 31;
    const int half  = lane >> 4;
    const int l15   = lane & 15;
    const int n = gate * HH + jtile * 16 + l15;        // gate column (B side)

    __shared__ __attribute__((aligned(16))) float hl[16][HH];   // 32 KB
    __shared__ float gl[4][16][16];
    __shared__ float red[16][16];

    // ---- stage this mtile's 16 rows of h (coalesced float4) ----
#pragma unroll
    for (int q = 0; q < 16; ++q) {
        const int e4 = threadIdx.x + q * 128;          // float4 index, 0..2047
        const int r  = e4 >> 7;                        // 128 float4 per row
        const int c  = (e4 & 127) * 4;
        *(float4*)&hl[r][c] = *(const float4*)&hin[(size_t)(mtile * 16 + r) * HH + c];
    }
    __syncthreads();

    v8f acc = {};
#pragma unroll 4
    for (int k0 = 0; k0 < HH; k0 += 4) {
        const int kk = k0 + half * 2;                  // f32 A 16x4 / B 4x16 layout
        v2f a2 = *(const v2f*)&hl[l15][kk];            // ds_load_b64
        v2f b2 = *(const v2f*)&Whh[(size_t)n * HH + kk];  // global_load_b64
        acc = __builtin_amdgcn_wmma_f32_16x16x4_f32(
                  false, a2, false, b2, (short)0, acc, false, false);
    }

#pragma unroll
    for (int r = 0; r < 8; ++r) {
        const int mm  = half ? (8 + r) : r;
        const int row = mtile * 16 + mm;               // batch
        const int col = gate * HH + jtile * 16 + l15;  // gate index
        gl[gate][mm][l15] = acc[r] + xW[((size_t)row * TT + t) * G4 + col];
    }
    __syncthreads();

    const float* Wa = Wfc;                             // W_a = Wfc[0..H)
#pragma unroll
    for (int p = 0; p < 2; ++p) {
        const int idx = threadIdx.x + p * 128;
        const int mm = idx >> 4, jj = idx & 15;
        const float ig = gl[0][mm][jj];
        const float fg = gl[1][mm][jj];
        const float gg = gl[2][mm][jj];
        const float og = gl[3][mm][jj];
        const int row  = mtile * 16 + mm;
        const int jcol = jtile * 16 + jj;
        const float cold = cst[row * HH + jcol];
        const float si = 1.f / (1.f + __expf(-ig));
        const float sf = 1.f / (1.f + __expf(-fg));
        const float so = 1.f / (1.f + __expf(-og));
        const float cn = sf * cold + si * tanhf(gg);
        const float hn = so * tanhf(cn);
        cst[row * HH + jcol]  = cn;
        hout[row * HH + jcol] = hn;
        red[mm][jj] = hn * Wa[jcol];
    }
    __syncthreads();
    if (threadIdx.x < 16) {
        float s = 0.f;
#pragma unroll
        for (int jj = 0; jj < 16; ++jj) s += red[threadIdx.x][jj];
        atomicAdd(aacc + (mtile * 16 + threadIdx.x) * TT + t, s);
    }
}

// ---------------------------------------------------------------------------
// Kernel 4: masked logsumexp loss -> scalar
// ---------------------------------------------------------------------------
__global__ void k_loss(const float* __restrict__ a,
                       const float* __restrict__ bp,
                       const float* __restrict__ bfc,
                       const int*  __restrict__ tlen,
                       float* __restrict__ out)
{
    const int tid = threadIdx.x;   // 256
    float lsum = 0.f, lcnt = 0.f;
    const float bf = bfc[0];
    for (int item = tid; item < BB * TT; item += 256) {
        const int b = item >> 6, i = item & 63;
        const int L = tlen[b];
        if (i > L - 2) continue;
        const float ai = a[item];
        float mx = -1e9f;
        for (int j = i + 1; j < L; ++j) mx = fmaxf(mx, ai + bp[b * TT + j] + bf);
        float s = 0.f;
        for (int j = 0; j < TT; ++j) {
            const bool cand = (j > i) && (j < L);
            const float v = cand ? (ai + bp[b * TT + j] + bf) : -1e9f;
            s += __expf(v - mx);
        }
        const float lse = mx + __logf(s);
        const float bnext = (i < TT - 1) ? bp[b * TT + i + 1] : 0.f;
        lsum += lse - (ai + bnext + bf);
        lcnt += 1.f;
    }
    __shared__ float rs[256], rc[256];
    rs[tid] = lsum; rc[tid] = lcnt;
    __syncthreads();
    for (int s = 128; s > 0; s >>= 1) {
        if (tid < s) { rs[tid] += rs[tid + s]; rc[tid] += rc[tid + s]; }
        __syncthreads();
    }
    if (tid == 0) out[0] = rs[0] / rc[0];
}

// ---------------------------------------------------------------------------
extern "C" void kernel_launch(void* const* d_in, const int* in_sizes, int n_in,
                              void* d_out, int out_size, void* d_ws, size_t ws_size,
                              hipStream_t stream) {
    const float* enc  = (const float*)d_in[0];
    const float* Wih  = (const float*)d_in[1];
    const float* Whh  = (const float*)d_in[2];
    const float* bih  = (const float*)d_in[3];
    const float* bhh  = (const float*)d_in[4];
    const float* Wfc  = (const float*)d_in[5];
    const float* bfc  = (const float*)d_in[6];
    const int*   ends = (const int*)d_in[7];
    const int*   tlen = (const int*)d_in[8];

    char* ws = (char*)d_ws;
    float* states = (float*)ws;                              //  8 MB (2048x1024 f32)
    float* xW     = (float*)(ws + (8u << 20));               // 16 MB (2048x2048 f32)
    float* h0     = (float*)(ws + (24u << 20));              // 64 KB
    float* h1     = h0 + BB * HH;                            // 64 KB
    float* cbuf   = h1 + BB * HH;                            // 64 KB
    float* bp     = cbuf + BB * HH;                          //  8 KB
    float* aacc   = bp + BB * TT;                            //  8 KB

    hipMemsetAsync(h0, 0, (size_t)3 * BB * HH * sizeof(float), stream);  // h0,h1,c
    hipMemsetAsync(aacc, 0, (size_t)BB * TT * sizeof(float), stream);

    k_states_bp<<<BB * TT, 256, 0, stream>>>(enc, ends, tlen, Wfc, states, bp);
    k_xw_gemm<<<128 * 32, 128, 0, stream>>>(states, Wih, bih, bhh, xW);
    for (int t = 0; t < TT; ++t) {
        float* hin  = (t & 1) ? h1 : h0;
        float* hout = (t & 1) ? h0 : h1;
        k_lstm_step<<<64, 128, 0, stream>>>(xW, Whh, Wfc, hin, hout, cbuf, aacc, t);
    }
    k_loss<<<1, 256, 0, stream>>>(aacc, bp, bfc, tlen, (float*)d_out);
}
